// LOIMLoss_1915555414355
// MI455X (gfx1250) — matrix operand order repaired and checked by reading, not compiled, not run on hardware
//
#include <hip/hip_runtime.h>

// ---------------------------------------------------------------------------
// LOIM focal loss on gfx1250 (MI455X).
// Pass 1: pack inputs/table to bf16 in a contiguous-k WMMA operand layout (d_ws).
// Pass 2: bf16 WMMA GEMM (16x16x32) with streaming softmax-sum, fixed max=30
//         (valid because all vectors are L2-normalized -> |logit| <= 30).
// Pass 3: combine partials, per-row label logit, focal loss, mean-reduce.
// ---------------------------------------------------------------------------

typedef __bf16 v16bf __attribute__((ext_vector_type(16)));
typedef float  v8f   __attribute__((ext_vector_type(8)));

#define NFEAT   256
#define NROWS   2048
#define NPIDS   5554
#define NCQ     50000
#define NCLASS  (NPIDS + NCQ)                    // 55554
#define OIM_SCALAR 30.0f
#define SOFT_MAX   30.0f                         // fixed softmax max bound
#define CSPLIT  64                               // class-dim split for partials
#define TILESC  ((NCLASS + 15) / 16)             // 3473 class tiles of 16
#define TPC     ((TILESC + CSPLIT - 1) / CSPLIT) // 55 tiles per chunk
#define RTILES  (NROWS / 16)                     // 128 row tiles

__device__ __forceinline__ v16bf cvt16(float4 f0, float4 f1, float4 f2, float4 f3) {
    v16bf t;
    t[0]=(__bf16)f0.x; t[1]=(__bf16)f0.y; t[2]=(__bf16)f0.z; t[3]=(__bf16)f0.w;
    t[4]=(__bf16)f1.x; t[5]=(__bf16)f1.y; t[6]=(__bf16)f1.z; t[7]=(__bf16)f1.w;
    t[8]=(__bf16)f2.x; t[9]=(__bf16)f2.y; t[10]=(__bf16)f2.z; t[11]=(__bf16)f2.w;
    t[12]=(__bf16)f3.x; t[13]=(__bf16)f3.y; t[14]=(__bf16)f3.z; t[15]=(__bf16)f3.w;
    return t;
}

// -------------------------------------------------------------------------
// Pack the class table into bf16 WMMA B-operand order:
// chunk index (ct*8 + kk)*32 + lane, 16 bf16 (32B) per chunk.
// k order used here (same as A): lane<16 col=lane K=k0..k0+15; lane>=16 K=k0+16..+31.
// -------------------------------------------------------------------------
__global__ __launch_bounds__(256)
void pack_b_kernel(const float* __restrict__ lut, const float* __restrict__ cq,
                   __bf16* __restrict__ bpack)
{
    const int ct   = blockIdx.x;                 // class tile (0..TILESC-1)
    const int kk   = threadIdx.x >> 5;           // k-step (0..7)
    const int lane = threadIdx.x & 31;
    const int cls  = ct * 16 + (lane & 15);
    const int kb   = kk * 32 + ((lane >= 16) ? 16 : 0);
    __bf16* dst = bpack + ((size_t)(ct * 8 + kk) * 32 + lane) * 16;

    v16bf t;
    if (cls < NCLASS) {
        const float* src = (cls < NPIDS) ? lut + (size_t)cls * NFEAT
                                         : cq  + (size_t)(cls - NPIDS) * NFEAT;
        t = cvt16(*(const float4*)(src + kb),     *(const float4*)(src + kb + 4),
                  *(const float4*)(src + kb + 8), *(const float4*)(src + kb + 12));
    } else {
#pragma unroll
        for (int i = 0; i < 16; ++i) t[i] = (__bf16)0.0f;
    }
    *(v16bf*)dst = t;
}

// -------------------------------------------------------------------------
// Pack inputs into bf16 WMMA A-operand order.
// k order used here (same as B): lane<16 row=lane, K=k0..k0+15;
//                           lane>=16 row=lane-16, K=k0+16..k0+31.
// -------------------------------------------------------------------------
__global__ __launch_bounds__(256)
void pack_a_kernel(const float* __restrict__ inputs, __bf16* __restrict__ apack)
{
    const int rt   = blockIdx.x;                 // row tile (0..127)
    const int kk   = threadIdx.x >> 5;
    const int lane = threadIdx.x & 31;
    const bool hi  = lane >= 16;
    const int kb0  = kk * 32 + (hi ? 16 : 0);
    const int kb1  = kb0 + 8;
    const float* src = inputs + (size_t)(rt * 16 + (lane & 15)) * NFEAT;
    v16bf t = cvt16(*(const float4*)(src + kb0), *(const float4*)(src + kb0 + 4),
                    *(const float4*)(src + kb1), *(const float4*)(src + kb1 + 4));
    *(v16bf*)(apack + ((size_t)(rt * 8 + kk) * 32 + lane) * 16) = t;
}

// -------------------------------------------------------------------------
// GEMM + streaming softmax sum. One wave32 per block.
// blockIdx.x = row tile, blockIdx.y = class chunk.
// -------------------------------------------------------------------------
__global__ __launch_bounds__(32)
void oim_partial_kernel(const __bf16* __restrict__ apack,
                        const __bf16* __restrict__ bpack,
                        float* __restrict__ psum)
{
    const int lane = threadIdx.x;
    const int rt   = blockIdx.x;
    const int cy   = blockIdx.y;
    const int n    = lane & 15;
    const bool hi  = lane >= 16;

    // Preload A tile: 8 k-step operands, straight 32B loads.
    v16bf a[8];
#pragma unroll
    for (int kk = 0; kk < 8; ++kk)
        a[kk] = *(const v16bf*)(apack + ((size_t)(rt * 8 + kk) * 32 + lane) * 16);

    float rsum[8];
#pragma unroll
    for (int j = 0; j < 8; ++j) rsum[j] = 0.0f;

    const int t0 = cy * TPC;
    const int t1 = (t0 + TPC < TILESC) ? (t0 + TPC) : TILESC;

    for (int ct = t0; ct < t1; ++ct) {
        const bool cvalid = (ct * 16 + n) < NCLASS;   // only false in tail tile
        const __bf16* bp = bpack + ((size_t)ct * 8 * 32 + lane) * 16;

        v8f acc = {0.f,0.f,0.f,0.f,0.f,0.f,0.f,0.f};
#pragma unroll
        for (int kk = 0; kk < 8; ++kk) {
            v16bf b = *(const v16bf*)(bp + (size_t)kk * 32 * 16);
            acc = __builtin_amdgcn_wmma_f32_16x16x32_bf16(
                      false, a[kk], false, b, (short)0, acc, false, false);
        }

        // logit = 30*acc; exp(logit - 30) = exp(30*acc - 30), branchless gate.
#pragma unroll
        for (int j = 0; j < 8; ++j) {
            const float e = __expf(fmaf(OIM_SCALAR, acc[j], -SOFT_MAX));
            rsum[j] += cvalid ? e : 0.0f;
        }
    }

    // Sum across the 16 lane-columns sharing each row.
#pragma unroll
    for (int m = 1; m < 16; m <<= 1) {
#pragma unroll
        for (int j = 0; j < 8; ++j)
            rsum[j] += __shfl_xor(rsum[j], m, 32);
    }

    if (n == 0) {
        const int rbase = rt * 16 + (hi ? 8 : 0);
#pragma unroll
        for (int j = 0; j < 8; ++j)
            psum[(size_t)(rbase + j) * CSPLIT + cy] = rsum[j];
    }
}

// -------------------------------------------------------------------------
// Combine partials; per-row label logit (bf16-rounded dot, matching the
// GEMM's precision); focal loss; mean reduction. Also label passthrough.
// -------------------------------------------------------------------------
__global__ __launch_bounds__(1024)
void oim_finish_kernel(const float* __restrict__ psum,
                       const float* __restrict__ inputs,
                       const float* __restrict__ lut,
                       const int*   __restrict__ targets,
                       float* __restrict__ out)
{
    __shared__ float red[1024];
    float local = 0.0f;

    for (int row = threadIdx.x; row < NROWS; row += 1024) {
        float s = 0.0f;
        for (int c = 0; c < CSPLIT; ++c)
            s += psum[(size_t)row * CSPLIT + c];

        const int lab = targets[row] - 1;
        float ce = 0.0f;
        if (lab != NPIDS) {                       // labeled (lab < NPIDS here)
            const float* x = inputs + (size_t)row * NFEAT;
            const float* w = lut + (size_t)lab * NFEAT;
            float dot = 0.0f;
            for (int k = 0; k < NFEAT; ++k)
                dot = fmaf((float)(__bf16)x[k], (float)(__bf16)w[k], dot);
            // logsumexp = SOFT_MAX + log(s);  ce = logsumexp - 30*dot
            ce = SOFT_MAX + __logf(s) - OIM_SCALAR * dot;
        }
        const float pt  = __expf(-ce);
        const float omp = 1.0f - pt;
        local += omp * omp * ce;                  // ALPHA=1, GAMMA=2
        out[1 + (size_t)NROWS * NFEAT + row] = (float)lab;
    }

    red[threadIdx.x] = local;
    __syncthreads();
#pragma unroll
    for (int st = 512; st > 0; st >>= 1) {
        if (threadIdx.x < st) red[threadIdx.x] += red[threadIdx.x + st];
        __syncthreads();
    }
    if (threadIdx.x == 0) out[0] = red[0] / (float)NROWS;
}

// -------------------------------------------------------------------------
extern "C" void kernel_launch(void* const* d_in, const int* in_sizes, int n_in,
                              void* d_out, int out_size, void* d_ws, size_t ws_size,
                              hipStream_t stream) {
    const float* inputs  = (const float*)d_in[0];
    const float* lut     = (const float*)d_in[1];
    const float* cq      = (const float*)d_in[2];
    const int*   targets = (const int*)  d_in[3];
    // d_in[4] = ious: unused by the reference computation.

    float* out = (float*)d_out;

    // Workspace layout (~30 MB total):
    __bf16* bpack = (__bf16*)d_ws;                                   // 28.44 MB
    __bf16* apack = bpack + (size_t)TILESC * 8 * 32 * 16;            //  1.00 MB
    float*  psum  = (float*)(apack + (size_t)RTILES * 8 * 32 * 16);  //  0.50 MB

    pack_b_kernel<<<dim3(TILESC), dim3(256), 0, stream>>>(lut, cq, bpack);
    pack_a_kernel<<<dim3(RTILES), dim3(256), 0, stream>>>(inputs, apack);

    oim_partial_kernel<<<dim3(RTILES, CSPLIT), dim3(32), 0, stream>>>(
        apack, bpack, psum);

    // inputs passthrough output (graph-capture-safe D2D copy).
    hipMemcpyAsync(out + 1, inputs, (size_t)NROWS * NFEAT * sizeof(float),
                   hipMemcpyDeviceToDevice, stream);

    oim_finish_kernel<<<dim3(1), dim3(1024), 0, stream>>>(
        psum, inputs, lut, targets, out);
}
